// ModelInteraction_37091337568315
// MI455X (gfx1250) — compile-verified, matrix operand order checked by reading, not run on hardware
//
#include <hip/hip_runtime.h>
#include <math.h>

typedef __attribute__((ext_vector_type(16))) _Float16 v16h;
typedef __attribute__((ext_vector_type(8)))  _Float16 v8h;
typedef __attribute__((ext_vector_type(8)))  float    v8f;
typedef __attribute__((ext_vector_type(4)))  unsigned int v4u;
typedef __attribute__((ext_vector_type(8)))  unsigned short v8us;

#define Bb   2
#define Nn   512
#define Mm   512
#define D0_  1024
#define Dd   100
#define Hc   50
#define KW_  7
#define ESTR 128   // padded row stride of e (halves)

#define SHUF16(lo, hi) __builtin_shufflevector((lo), (hi), 0,1,2,3,4,5,6,7,8,9,10,11,12,13,14,15)

static __device__ __forceinline__ v8h habs8(v8h x) {   // |x| via sign-bit mask
  v8us u;
  __builtin_memcpy(&u, &x, sizeof(u));
  u &= (unsigned short)0x7FFF;
  v8h r;
  __builtin_memcpy(&r, &u, sizeof(r));
  return r;
}

// ---------------------------------------------------------------------------
// Kernel 0: prepack all f16 operand banks (one elementwise pass).
//   xh  [2048][1024]: f16 copy of [x0;x1]
//   Wt  [128][1024] : Wt[col][c]   = W_proj[c][col]     (col>=100 -> 0)
//   WhT [2][64][128]: WhT[p][h][c] = W_hid[p*100+c][h]  (h>=50 | c>=100 -> 0)
// ---------------------------------------------------------------------------
__global__ __launch_bounds__(256) void k0_pack(const float* __restrict__ x0,
                                               const float* __restrict__ x1,
                                               const float* __restrict__ Wp,
                                               const float* __restrict__ Whid,
                                               _Float16* __restrict__ xh,
                                               _Float16* __restrict__ Wt,
                                               _Float16* __restrict__ WhT) {
  const int NX = Bb * Nn * D0_;              // 1048576
  int gid = blockIdx.x * 256 + threadIdx.x;
  if (gid < 2 * NX) {
    float v = (gid < NX) ? x0[gid] : x1[gid - NX];
    xh[gid] = (_Float16)v;
  }
  int g1 = gid - 2 * NX;
  if (g1 >= 0 && g1 < 128 * 1024) {
    int col = g1 >> 10, c = g1 & 1023;
    Wt[g1] = (col < Dd) ? (_Float16)Wp[(size_t)c * Dd + col] : (_Float16)0.f;
  }
  int g2 = g1 - 128 * 1024;
  if (g2 >= 0 && g2 < 2 * 64 * 128) {
    int p = g2 >> 13, h = (g2 >> 7) & 63, c = g2 & 127;
    WhT[g2] = (h < Hc && c < Dd) ? (_Float16)Whid[((size_t)p * Dd + c) * Hc + h]
                                 : (_Float16)0.f;
  }
}

// ---------------------------------------------------------------------------
// Kernel 1: e = relu([x0;x1] @ W_proj + b_proj) -> padded f16 rows [2048 x 128].
// Block = 8 waves sharing one 16-col tile (Wt slice in 32KB LDS).
// Inner loop: 2 global b128 + 2 ds b128 + 1 WMMA, no converts.
// ---------------------------------------------------------------------------
__global__ __launch_bounds__(256) void k1_embed(const _Float16* __restrict__ xh,
                                                const _Float16* __restrict__ Wt,
                                                const float* __restrict__ bp,
                                                _Float16* __restrict__ e) {
  __shared__ __attribute__((aligned(16))) _Float16 wsl[16 * 1024]; // 32KB
  int tid = threadIdx.x;
  int bid = blockIdx.x;            // 128 blocks
  int ct  = bid & 7;               // col tile
  int rtg = bid >> 3;              // row-tile group (8 row tiles/block)

  { // stage Wt slice (coalesced b128 copy)
    const v4u* src = (const v4u*)(Wt + (size_t)ct * 16 * 1024);
    v4u* dst = (v4u*)wsl;
    for (int i = tid; i < 16 * 1024 / 8; i += 256) dst[i] = src[i];
  }
  __syncthreads();

  int w = tid >> 5, lane = tid & 31, lr = lane & 15, kh = lane >> 4;
  int rt = rtg * 8 + w;
  int arow = rt * 16 + lr;
  const _Float16* rp   = xh + (size_t)arow * D0_;
  const _Float16* wrow = wsl + lr * 1024;

  v8f acc = {};
  for (int ks = 0; ks < D0_ / 32; ++ks) {
    int base = ks * 32 + kh * 8;
    v16h a  = SHUF16(*(const v8h*)(rp + base),   *(const v8h*)(rp + base + 16));
    v16h bf = SHUF16(*(const v8h*)(wrow + base), *(const v8h*)(wrow + base + 16));
    acc = __builtin_amdgcn_wmma_f32_16x16x32_f16(false, a, false, bf,
                                                 (short)0, acc, false, false);
  }

  int col = ct * 16 + lr;
  bool colv = (col < Dd);
  float bias = bp[colv ? col : 0];          // clamped-address load, no exec branch
  if (!colv) bias = 0.f;
  int crow = rt * 16 + kh * 8;              // C layout: M = r + 8*(lane>=16)
#pragma unroll
  for (int r = 0; r < 8; ++r) {
    float v = acc[r] + bias;
    v = v > 0.f ? v : 0.f;
    e[(size_t)(crow + r) * ESTR + col] = (_Float16)v;  // padded cols -> exact 0
  }
}

// ---------------------------------------------------------------------------
// Kernel 2: Hh = relu(Z @ W_hid + b_hid) via two K=128 WMMA chains
// (diff + mul parts) into one accumulator. Wave = (b, m, n-tile).
// Interaction built in packed f16: v_pk_add/mul_f16 + v_and_b32 (abs).
// WhT (32KB) + e1 row + bias staged in LDS. 32 WMMAs per wave.
// ---------------------------------------------------------------------------
__global__ __launch_bounds__(256) void k2_inter(const _Float16* __restrict__ e,
                                                const _Float16* __restrict__ WhT,
                                                const float* __restrict__ bh,
                                                _Float16* __restrict__ HhT) {
  __shared__ __attribute__((aligned(16))) _Float16 whs[2 * 64 * 128]; // 32KB
  __shared__ __attribute__((aligned(16))) _Float16 e1s[ESTR];
  __shared__ float bhs[64];
  int tid = threadIdx.x;
  int bid = blockIdx.x;

  { // stage WhT: 2048 x b128
    const v4u* src = (const v4u*)WhT;
    v4u* dst = (v4u*)whs;
    for (int i = tid; i < 2 * 64 * 128 / 8; i += 256) dst[i] = src[i];
  }
  int wave0 = bid * 8;                 // 8 waves per block share (b, m)
  int b = wave0 / (Mm * 32);
  int rem0 = wave0 % (Mm * 32);
  int m = rem0 / 32;
  if (tid < ESTR) e1s[tid] = e[((size_t)(Bb * Nn + b * Mm + m)) * ESTR + tid];
  if (tid < 64)   bhs[tid] = (tid < Hc) ? bh[tid] : 0.f;
  __syncthreads();

  int w = tid >> 5, lane = tid & 31, lr = lane & 15, kh = lane >> 4;
  int nt = (rem0 % 32) + w;
  const _Float16* e0r = e + ((size_t)b * Nn + nt * 16 + lr) * ESTR;

  v8f acc[4] = {};
  for (int ks = 0; ks < 4; ++ks) {
    int base = ks * 32 + kh * 8;
    v8h p0 = *(const v8h*)(e0r + base);
    v8h p1 = *(const v8h*)(e0r + base + 16);
    v8h q0 = *(const v8h*)(e1s + base);
    v8h q1 = *(const v8h*)(e1s + base + 16);
    v16h zd = SHUF16(habs8(p0 - q0), habs8(p1 - q1));
    v16h zm = SHUF16(p0 * q0, p1 * q1);
#pragma unroll
    for (int t = 0; t < 4; ++t) {
      int h = t * 16 + lr;
      const _Float16* wr0 = whs + (size_t)h * 128 + base;          // diff part
      const _Float16* wr1 = whs + (size_t)(64 + h) * 128 + base;   // mul part
      v16h ad = SHUF16(*(const v8h*)wr0, *(const v8h*)(wr0 + 16));
      acc[t] = __builtin_amdgcn_wmma_f32_16x16x32_f16(false, ad, false, zd,
                                                      (short)0, acc[t], false, false);
      v16h am = SHUF16(*(const v8h*)wr1, *(const v8h*)(wr1 + 16));
      acc[t] = __builtin_amdgcn_wmma_f32_16x16x32_f16(false, am, false, zm,
                                                      (short)0, acc[t], false, false);
    }
  }

  int n = nt * 16 + lr;                 // D: N = lane&15 -> coalesced n
#pragma unroll
  for (int t = 0; t < 4; ++t) {
#pragma unroll
    for (int r = 0; r < 8; ++r) {
      int h = t * 16 + r + 8 * kh;      // M = r + 8*(lane>=16)
      if (h < Hc) {                     // constant-folds true for t<3
        float v = acc[t][r] + bhs[h];
        v = v > 0.f ? v : 0.f;
        HhT[(((size_t)b * Hc + h) * Mm + m) * Nn + n] = (_Float16)v;
      }
    }
  }
}

// ---------------------------------------------------------------------------
// Kernel 3: 7x7 SAME conv (50ch) + bias + sigmoid + analytic W(n,m);
// writes yhat[b][m][n] + per-block (sum,sumsq) partials.
// ---------------------------------------------------------------------------
__global__ __launch_bounds__(256) void k3_conv(const _Float16* __restrict__ HhT,
                                               const float* __restrict__ cw,
                                               const float* __restrict__ cb,
                                               const float* __restrict__ theta,
                                               const float* __restrict__ lam,
                                               float* __restrict__ yhat,
                                               float* __restrict__ part) {
  __shared__ float wsm[Hc * KW_ * KW_];
  __shared__ float tile[22 * 24];
  __shared__ float red1[256];
  __shared__ float red2[256];

  int tid = threadIdx.x;
  int bid = blockIdx.x;              // ((b*32 + mt)*32 + nt)
  int b  = bid >> 10;
  int mt = (bid >> 5) & 31;
  int nt = bid & 31;
  int tx = tid & 15, ty = tid >> 4;
  int mbase = mt * 16, nbase = nt * 16;

  for (int i = tid; i < Hc * KW_ * KW_; i += 256) wsm[i] = cw[i];
  __syncthreads();

  float accv = 0.f;
  for (int h = 0; h < Hc; ++h) {
    const _Float16* plane = HhT + (((size_t)b * Hc + h) * Mm) * Nn;
    for (int i = tid; i < 22 * 22; i += 256) {
      int rr = i / 22, cc = i % 22;
      int mg = mbase - 3 + rr, ng = nbase - 3 + cc;
      bool inb = (mg >= 0 && mg < Mm && ng >= 0 && ng < Nn);
      int mgc = mg < 0 ? 0 : (mg > Mm - 1 ? Mm - 1 : mg);
      int ngc = ng < 0 ? 0 : (ng > Nn - 1 ? Nn - 1 : ng);
      float v = (float)plane[(size_t)mgc * Nn + ngc];   // always load (clamped)
      tile[rr * 24 + cc] = inb ? v : 0.f;               // select, no branch
    }
    __syncthreads();
    const float* wh = &wsm[h * 49];
#pragma unroll
    for (int dn = 0; dn < 7; ++dn)
#pragma unroll
      for (int dm = 0; dm < 7; ++dm)
        accv += tile[(ty + dm) * 24 + (tx + dn)] * wh[dn * 7 + dm];
    __syncthreads();
  }

  int n = nbase + tx, m = mbase + ty;
  float sig = 1.f / (1.f + expf(-(accv + cb[0])));
  float th = theta[0], la = lam[0];
  float an = ((float)n - 255.5f) / 256.5f;
  float bm = ((float)m - 255.5f) / 256.5f;
  float Wnm = (1.f - th) * expf(-la * an * an) * expf(-la * bm * bm) + th;
  float y = sig * Wnm;
  yhat[((size_t)b * Mm + m) * Nn + n] = y;

  red1[tid] = y; red2[tid] = y * y;
  __syncthreads();
  for (int s = 128; s > 0; s >>= 1) {
    if (tid < s) { red1[tid] += red1[tid + s]; red2[tid] += red2[tid + s]; }
    __syncthreads();
  }
  if (tid == 0) { part[bid * 2] = red1[0]; part[bid * 2 + 1] = red2[0]; }
}

// ---------------------------------------------------------------------------
// Kernel 4: deterministic finalize per batch (fixed-order tree reductions).
// ---------------------------------------------------------------------------
__global__ __launch_bounds__(256) void k4_final(const float* __restrict__ yhat,
                                                const float* __restrict__ part,
                                                const float* __restrict__ gamma,
                                                float* __restrict__ out) {
  __shared__ float r1[256], r2[256];
  __shared__ float sthr;
  int tid = threadIdx.x;
  int b = blockIdx.x;

  float s = 0.f, s2 = 0.f;
  for (int i = tid; i < 1024; i += 256) {
    s  += part[(b * 1024 + i) * 2];
    s2 += part[(b * 1024 + i) * 2 + 1];
  }
  r1[tid] = s; r2[tid] = s2;
  __syncthreads();
  for (int st = 128; st > 0; st >>= 1) {
    if (tid < st) { r1[tid] += r1[tid + st]; r2[tid] += r2[tid + st]; }
    __syncthreads();
  }
  if (tid == 0) {
    float nm = (float)(Mm * Nn);
    float mu  = r1[0] / nm;
    float var = (r2[0] - r1[0] * r1[0] / nm) / (nm - 1.f);
    sthr = mu + gamma[0] * var;
  }
  __syncthreads();
  float thr = sthr;
  float qs = 0.f, cnt = 0.f;
  const float* yb = yhat + (size_t)b * Mm * Nn;
  for (int i = tid; i < Mm * Nn; i += 256) {
    float q = yb[i] - thr;
    if (q > 0.f) { qs += q; cnt += 1.f; }
  }
  r1[tid] = qs; r2[tid] = cnt;
  __syncthreads();
  for (int st = 128; st > 0; st >>= 1) {
    if (tid < st) { r1[tid] += r1[tid + st]; r2[tid] += r2[tid + st]; }
    __syncthreads();
  }
  if (tid == 0) {
    float phat = r1[0] / (r2[0] + 1.f);
    float v = 1.f / (1.f + expf(-20.f * (phat - 0.5f)));
    v = v < 0.f ? 0.f : (v > 1.f ? 1.f : v);
    out[b] = v;
  }
}

// ---------------------------------------------------------------------------
extern "C" void kernel_launch(void* const* d_in, const int* in_sizes, int n_in,
                              void* d_out, int out_size, void* d_ws, size_t ws_size,
                              hipStream_t stream) {
  const float* x0    = (const float*)d_in[0];
  const float* x1    = (const float*)d_in[1];
  const float* Wp    = (const float*)d_in[2];
  const float* bp    = (const float*)d_in[3];
  const float* Whid  = (const float*)d_in[4];
  const float* bh    = (const float*)d_in[5];
  const float* cw    = (const float*)d_in[6];
  const float* cb    = (const float*)d_in[7];
  const float* theta = (const float*)d_in[8];
  const float* lam   = (const float*)d_in[9];
  const float* gamma = (const float*)d_in[10];

  // ws layout (~56.8 MB, all offsets 16B+ aligned):
  char* ws = (char*)d_ws;
  size_t offX  = 0;                                    // 2048*1024 f16 = 4 MB
  size_t offWt = offX  + (size_t)2048 * 1024 * 2;      // 128*1024 f16  = 256 KB
  size_t offWh = offWt + (size_t)128 * 1024 * 2;       // 2*64*128 f16  =  32 KB
  size_t offE  = offWh + (size_t)2 * 64 * 128 * 2;     // 2048*128 f16  = 512 KB
  size_t offH  = offE  + (size_t)2048 * ESTR * 2;      // 2*50*512*512 f16 = 50 MB
  size_t offY  = offH  + (size_t)Bb * Hc * Mm * Nn * 2;
  size_t offP  = offY  + (size_t)Bb * Mm * Nn * 4;
  _Float16* xh   = (_Float16*)(ws + offX);
  _Float16* Wt   = (_Float16*)(ws + offWt);
  _Float16* WhT  = (_Float16*)(ws + offWh);
  _Float16* e    = (_Float16*)(ws + offE);
  _Float16* HhT  = (_Float16*)(ws + offH);
  float*    yhat = (float*)(ws + offY);
  float*    part = (float*)(ws + offP);

  k0_pack<<<8768, 256, 0, stream>>>(x0, x1, Wp, Whid, xh, Wt, WhT);
  k1_embed<<<128, 256, 0, stream>>>(xh, Wt, bp, e);     // 1024 waves
  k2_inter<<<4096, 256, 0, stream>>>(e, WhT, bh, HhT);  // 32768 waves
  k3_conv<<<2048, 256, 0, stream>>>(HhT, cw, cb, theta, lam, yhat, part);
  k4_final<<<2, 256, 0, stream>>>(yhat, part, gamma, (float*)d_out);
}
